// COBRAGatingUnit_89850715832421
// MI455X (gfx1250) — matrix operand order repaired and checked by reading, not compiled, not run on hardware
//
#include <hip/hip_runtime.h>
#include <math.h>

// ---------------------------------------------------------------------------
// Types for CDNA5 WMMA (wave32): v_wmma_f32_16x16x32_bf16
// ---------------------------------------------------------------------------
typedef __bf16 bf16;
typedef __attribute__((ext_vector_type(16))) __bf16 v16bf;
typedef __attribute__((ext_vector_type(8)))  float  v8f;
typedef __attribute__((ext_vector_type(4)))  int    v4i;

union FragBF16 {
    v16bf v;
    unsigned int u[8];
};

#define D_MODEL   768
#define D_INNER   1536
#define D_XZ      3072
#define D_STATE   16
#define DT_RANK   48
#define DT_PAD    64
#define SEQ       1024
#define BATCH     2
#define ROWS      2048   // BATCH * SEQ
#define XPROJ_N   80

// gfx1250 async global->LDS path (ASYNCcnt-tracked), guarded so compile never breaks
#if defined(__has_builtin)
#  if __has_builtin(__builtin_amdgcn_global_load_async_to_lds_b128)
#    define HAVE_ASYNC_LDS 1
#  endif
#  if __has_builtin(__builtin_amdgcn_s_wait_asynccnt)
#    define HAVE_WAIT_ASYNC 1
#  endif
#endif

#define AS1 __attribute__((address_space(1)))
#define AS3 __attribute__((address_space(3)))

// ---------------------------------------------------------------------------
// Weight convert: W (K x N, f32, row-major) -> Bt (N x Kpad, bf16), zero-pad K
// ---------------------------------------------------------------------------
__global__ void convert_wt_kernel(const float* __restrict__ W, bf16* __restrict__ Bt,
                                  int K, int N, int Kpad) {
    size_t i = (size_t)blockIdx.x * 256 + threadIdx.x;
    size_t total = (size_t)N * Kpad;
    if (i >= total) return;
    int k = (int)(i % Kpad);
    int n = (int)(i / Kpad);
    float v = (k < K) ? W[(size_t)k * N + n] : 0.0f;
    Bt[i] = (bf16)v;
}

// ---------------------------------------------------------------------------
// RMSNorm over D_MODEL, emit bf16. One block (256 thr) per row; 3 elems/thread.
// ---------------------------------------------------------------------------
__global__ void rmsnorm_kernel(const float* __restrict__ x, const float* __restrict__ w,
                               bf16* __restrict__ xn) {
    int row = blockIdx.x;
    int tid = threadIdx.x;
    const float* xr = x + (size_t)row * D_MODEL;
    float v0 = xr[tid], v1 = xr[tid + 256], v2 = xr[tid + 512];
    float s = v0 * v0 + v1 * v1 + v2 * v2;
    #pragma unroll
    for (int off = 16; off > 0; off >>= 1) s += __shfl_xor(s, off, 32);
    __shared__ float red[8];
    if ((tid & 31) == 0) red[tid >> 5] = s;
    __syncthreads();
    if (tid == 0) {
        float t = 0.f;
        #pragma unroll
        for (int i = 0; i < 8; ++i) t += red[i];
        red[0] = rsqrtf(t * (1.0f / (float)D_MODEL) + 1e-5f);
    }
    __syncthreads();
    float scale = red[0];
    bf16* o = xn + (size_t)row * D_MODEL;
    o[tid]       = (bf16)(v0 * scale * w[tid]);
    o[tid + 256] = (bf16)(v1 * scale * w[tid + 256]);
    o[tid + 512] = (bf16)(v2 * scale * w[tid + 512]);
}

// ---------------------------------------------------------------------------
// LDS-staged WMMA GEMM: C(MxN,f32) = A(MxK,bf16) x Bt(NxK,bf16)^T [+ addsrc]
// Block = 256 threads = 8 waves, block tile 64(M) x 64(N), K-step 32.
// Waves arranged 2(M) x 4(N); each wave computes 32x16 (two 16x16x32 WMMAs).
// A/B tiles double-buffered in LDS; staged with async global->LDS when
// available (ASYNCcnt pipeline: wait own asyncs, barrier, issue k+1, compute).
// Requires: M % 64 == 0, N % 64 == 0, K % 32 == 0.
// ---------------------------------------------------------------------------
__global__ __launch_bounds__(256)
void wmma_gemm_lds_kernel(const bf16* __restrict__ A, const bf16* __restrict__ Bt,
                          float* __restrict__ C, const float* __restrict__ addsrc,
                          int M, int N, int K) {
    __shared__ bf16 sA[2][64 * 32];   // [m_local][k_local]
    __shared__ bf16 sB[2][64 * 32];   // [n_local][k_local]

    int bn_count = N >> 6;
    int bm = blockIdx.x / bn_count;
    int bn = blockIdx.x - bm * bn_count;
    if (bm >= (M >> 6)) return;
    int m_blk = bm << 6;
    int n_blk = bn << 6;

    int tid = threadIdx.x;

    // Staging: 64 rows x 4 chunks of 8 bf16 (16B) = 256 thread-chunks per tile
    int srow = tid >> 2;            // 0..63
    int skk  = (tid & 3) * 8;       // 0,8,16,24
    const bf16* gA = A  + (size_t)(m_blk + srow) * K + skk;
    const bf16* gB = Bt + (size_t)(n_blk + srow) * K + skk;
    int soff = srow * 32 + skk;

    auto stage = [&](int buf, int k0) {
#if defined(HAVE_ASYNC_LDS)
        __builtin_amdgcn_global_load_async_to_lds_b128(
            (AS1 v4i*)(gA + k0), (AS3 v4i*)&sA[buf][soff], 0, 0);
        __builtin_amdgcn_global_load_async_to_lds_b128(
            (AS1 v4i*)(gB + k0), (AS3 v4i*)&sB[buf][soff], 0, 0);
#else
        *(uint4*)&sA[buf][soff] = *(const uint4*)(gA + k0);
        *(uint4*)&sB[buf][soff] = *(const uint4*)(gB + k0);
#endif
    };

    // Per-wave tile coordinates
    int wave = tid >> 5;
    int lane = tid & 31;
    int wm0 = (wave >> 2) << 5;     // 0 or 32 (local M base; two 16-row subtiles)
    int wn0 = (wave & 3) << 4;      // 0,16,32,48 (local N base)

    // WMMA lane layouts (16-bit A 16x32; 16-bit B 32x16; f32 C 16x16)
    int arow = lane & 15;
    int akb  = (lane < 16) ? 0 : 8;
    int bcol = lane & 15;
    int bkb  = (lane < 16) ? 0 : 8;

    v8f acc0 = {};
    v8f acc1 = {};

    stage(0, 0);
    int cur = 0;
    for (int k0 = 0; k0 < K; k0 += 32) {
#if defined(HAVE_WAIT_ASYNC)
        __builtin_amdgcn_s_wait_asynccnt(0);
#endif
        __syncthreads();            // buf[cur] ready; prev reads of buf[cur^1] done
        if (k0 + 32 < K) stage(cur ^ 1, k0 + 32);

        const bf16* aptr0 = &sA[cur][(wm0 + arow) * 32 + akb];
        const bf16* aptr1 = aptr0 + 16 * 32;
        const bf16* bptr  = &sB[cur][(wn0 + bcol) * 32 + bkb];

        FragBF16 a0, a1, b;
        *(uint4*)&a0.u[0] = *(const uint4*)(aptr0);        // k = akb .. akb+7
        *(uint4*)&a0.u[4] = *(const uint4*)(aptr0 + 16);   // k = akb+16 .. akb+23
        *(uint4*)&a1.u[0] = *(const uint4*)(aptr1);
        *(uint4*)&a1.u[4] = *(const uint4*)(aptr1 + 16);
        *(uint4*)&b.u[0]  = *(const uint4*)(bptr);         // k = bkb .. bkb+7
        *(uint4*)&b.u[4]  = *(const uint4*)(bptr + 16);    // k = bkb+16 .. bkb+23

        acc0 = __builtin_amdgcn_wmma_f32_16x16x32_bf16(false, a0.v, false, b.v,
                                                       (short)0, acc0, false, false);
        acc1 = __builtin_amdgcn_wmma_f32_16x16x32_bf16(false, a1.v, false, b.v,
                                                       (short)0, acc1, false, false);
        cur ^= 1;
    }

    // C layout: VGPR r -> lanes 0-15 M=r, lanes 16-31 M=8+r; N = lane&15
    int nn = n_blk + wn0 + (lane & 15);
    int mbase = m_blk + wm0 + ((lane >> 4) << 3);
    #pragma unroll
    for (int r = 0; r < 8; ++r) {
        int mr = mbase + r;
        float c0 = acc0[r];
        float c1 = acc1[r];
        if (addsrc) {
            c0 += addsrc[(size_t)mr * N + nn];
            c1 += addsrc[(size_t)(mr + 16) * N + nn];
        }
        C[(size_t)mr * N + nn]        = c0;
        C[(size_t)(mr + 16) * N + nn] = c1;
    }
}

// ---------------------------------------------------------------------------
// Direct-fragment WMMA GEMM (for small N not a multiple of 64, e.g. N=80).
// Each wave computes a 32(M) x 16(N) tile. M%32==0, N%16==0, K%32==0.
// ---------------------------------------------------------------------------
__global__ void wmma_gemm_kernel(const bf16* __restrict__ A, const bf16* __restrict__ Bt,
                                 float* __restrict__ C, const float* __restrict__ addsrc,
                                 int M, int N, int K) {
    int wid = blockIdx.x * 8 + (threadIdx.x >> 5);
    int tiles_n = N >> 4;
    int tiles_m = M >> 5;
    int tm = wid / tiles_n;
    int tn = wid - tm * tiles_n;
    if (tm >= tiles_m) return;
    int m0 = tm << 5;
    int n0 = tn << 4;

    int lane = threadIdx.x & 31;
    int arow = lane & 15;
    int akb  = (lane < 16) ? 0 : 8;
    int bcol = lane & 15;
    int bkb  = (lane < 16) ? 0 : 8;

    const bf16* Arow0 = A + (size_t)(m0 + arow) * K;
    const bf16* Arow1 = Arow0 + (size_t)16 * K;
    const bf16* Brow  = Bt + (size_t)(n0 + bcol) * K;

    v8f acc0 = {};
    v8f acc1 = {};

    for (int k0 = 0; k0 < K; k0 += 32) {
        __builtin_prefetch(Arow0 + k0 + 128, 0, 0);
        FragBF16 a0, a1, b;
        #pragma unroll
        for (int i = 0; i < 8; ++i) {
            int ka = k0 + akb + ((i < 4) ? (2 * i) : (2 * i + 8));
            a0.u[i] = *(const unsigned int*)(Arow0 + ka);
            a1.u[i] = *(const unsigned int*)(Arow1 + ka);
            int kb = k0 + bkb + ((i < 4) ? (2 * i) : (2 * i + 8));
            b.u[i]  = *(const unsigned int*)(Brow + kb);
        }
        acc0 = __builtin_amdgcn_wmma_f32_16x16x32_bf16(false, a0.v, false, b.v,
                                                       (short)0, acc0, false, false);
        acc1 = __builtin_amdgcn_wmma_f32_16x16x32_bf16(false, a1.v, false, b.v,
                                                       (short)0, acc1, false, false);
    }

    int nn = n0 + (lane & 15);
    int mbase = m0 + ((lane >> 4) << 3);
    #pragma unroll
    for (int r = 0; r < 8; ++r) {
        int mr = mbase + r;
        float c0 = acc0[r];
        float c1 = acc1[r];
        if (addsrc) {
            c0 += addsrc[(size_t)mr * N + nn];
            c1 += addsrc[(size_t)(mr + 16) * N + nn];
        }
        C[(size_t)mr * N + nn]        = c0;
        C[(size_t)(mr + 16) * N + nn] = c1;
    }
}

// ---------------------------------------------------------------------------
// Depthwise causal conv (D_CONV=4) over xi = xz[:, :D_INNER], + bias, SiLU.
// ---------------------------------------------------------------------------
__global__ void conv_silu_kernel(const float* __restrict__ xz, const float* __restrict__ cw,
                                 const float* __restrict__ cb, float* __restrict__ xc,
                                 bf16* __restrict__ xcb) {
    size_t i = (size_t)blockIdx.x * 256 + threadIdx.x;
    if (i >= (size_t)ROWS * D_INNER) return;
    int d   = (int)(i % D_INNER);
    int row = (int)(i / D_INNER);
    int t = row & (SEQ - 1);
    int b = row >> 10;
    float acc = cb[d];
    #pragma unroll
    for (int j = 0; j < 4; ++j) {
        int tt = t - 3 + j;
        if (tt >= 0)
            acc += cw[d * 4 + j] * xz[(size_t)(b * SEQ + tt) * D_XZ + d];
    }
    float s = acc / (1.0f + __expf(-acc));   // SiLU
    xc[i]  = s;
    xcb[i] = (bf16)s;
}

// ---------------------------------------------------------------------------
// Split dbc (ROWS x 80): cols 0..47 -> dlt (bf16, padded to 64 w/ zeros),
// 48..63 -> Bm (f32), 64..79 -> Cm (f32). 96 threads per row.
// ---------------------------------------------------------------------------
__global__ void extract_dbc_kernel(const float* __restrict__ dbc, bf16* __restrict__ dlt,
                                   float* __restrict__ Bm, float* __restrict__ Cm) {
    size_t i = (size_t)blockIdx.x * 256 + threadIdx.x;
    if (i >= (size_t)ROWS * 96) return;
    int c   = (int)(i % 96);
    int row = (int)(i / 96);
    if (c < DT_RANK) {
        dlt[(size_t)row * DT_PAD + c] = (bf16)dbc[(size_t)row * XPROJ_N + c];
    } else if (c < DT_RANK + D_STATE) {
        dlt[(size_t)row * DT_PAD + c] = (bf16)0.0f;                  // zero pad K
        Bm[(size_t)row * D_STATE + (c - DT_RANK)] = dbc[(size_t)row * XPROJ_N + c];
    } else if (c < XPROJ_N) {
        Cm[(size_t)row * D_STATE + (c - DT_RANK - D_STATE)] = dbc[(size_t)row * XPROJ_N + c];
    }
}

// ---------------------------------------------------------------------------
// delta = softplus(delta_pre + dt_b), in place
// ---------------------------------------------------------------------------
__global__ void delta_softplus_kernel(float* __restrict__ delta, const float* __restrict__ dtb) {
    size_t i = (size_t)blockIdx.x * 256 + threadIdx.x;
    if (i >= (size_t)ROWS * D_INNER) return;
    int d = (int)(i % D_INNER);
    float v = delta[i] + dtb[d];
    delta[i] = (v > 20.0f) ? v : log1pf(__expf(v));
}

// ---------------------------------------------------------------------------
// Selective scan: sequential over t, parallel over (batch, d) channels.
// h[16] in registers; B/C broadcast via double-buffered LDS (1 barrier/step).
// ---------------------------------------------------------------------------
__global__ void selective_scan_kernel(const float* __restrict__ xz, const float* __restrict__ xc,
                                      const float* __restrict__ delta,
                                      const float* __restrict__ Bm, const float* __restrict__ Cm,
                                      const float* __restrict__ A_log, const float* __restrict__ Dp,
                                      bf16* __restrict__ yg) {
    const int chunks = D_INNER / 256;           // 6
    int b = blockIdx.x / chunks;
    int d = (blockIdx.x % chunks) * 256 + threadIdx.x;

    float A[D_STATE];
    #pragma unroll
    for (int n = 0; n < D_STATE; ++n) A[n] = -__expf(A_log[(size_t)d * D_STATE + n]);
    float Dd = Dp[d];
    float h[D_STATE];
    #pragma unroll
    for (int n = 0; n < D_STATE; ++n) h[n] = 0.0f;

    __shared__ float sB[2][D_STATE];
    __shared__ float sC[2][D_STATE];

    for (int t = 0; t < SEQ; ++t) {
        size_t row = (size_t)b * SEQ + t;
        int buf = t & 1;
        if (threadIdx.x < D_STATE)
            sB[buf][threadIdx.x] = Bm[row * D_STATE + threadIdx.x];
        else if (threadIdx.x < 2 * D_STATE)
            sC[buf][threadIdx.x - D_STATE] = Cm[row * D_STATE + (threadIdx.x - D_STATE)];
        __syncthreads();

        float dv = delta[row * D_INNER + d];
        float xv = xc[row * D_INNER + d];
        float dx = dv * xv;
        float y = 0.0f;
        #pragma unroll
        for (int n = 0; n < D_STATE; ++n) {
            h[n] = __expf(dv * A[n]) * h[n] + dx * sB[buf][n];
            y += h[n] * sC[buf][n];
        }
        y += Dd * xv;
        float z = xz[row * D_XZ + D_INNER + d];
        float g = z / (1.0f + __expf(-z));      // SiLU(z)
        yg[row * D_INNER + d] = (bf16)(y * g);
    }
}

// ---------------------------------------------------------------------------
// out = u * v
// ---------------------------------------------------------------------------
__global__ void final_mul_kernel(const float* __restrict__ u, const float* __restrict__ v,
                                 float* __restrict__ out) {
    size_t i = (size_t)blockIdx.x * 256 + threadIdx.x;
    if (i >= (size_t)ROWS * D_MODEL) return;
    out[i] = u[i] * v[i];
}

// ---------------------------------------------------------------------------
// Host launcher
// ---------------------------------------------------------------------------
extern "C" void kernel_launch(void* const* d_in, const int* in_sizes, int n_in,
                              void* d_out, int out_size, void* d_ws, size_t ws_size,
                              hipStream_t stream) {
    (void)in_sizes; (void)n_in; (void)out_size; (void)ws_size;
    const float* x = (const float*)d_in[0];

    // Workspace bump allocator (256B aligned)
    char* wp = (char*)d_ws;
    auto alloc = [&](size_t bytes) -> void* {
        void* r = (void*)wp;
        wp += (bytes + 255) & ~(size_t)255;
        return r;
    };

    bf16*  wi_t  = (bf16*) alloc((size_t)D_XZ   * D_MODEL * sizeof(bf16));  // in_proj^T
    bf16*  wx_t  = (bf16*) alloc((size_t)XPROJ_N* D_INNER * sizeof(bf16));  // x_proj^T
    bf16*  wd_t  = (bf16*) alloc((size_t)D_INNER* DT_PAD  * sizeof(bf16));  // dt_w^T padded
    bf16*  wo_t  = (bf16*) alloc((size_t)D_MODEL* D_INNER * sizeof(bf16));  // out_proj^T
    bf16*  xn    = (bf16*) alloc((size_t)ROWS * D_MODEL * sizeof(bf16));
    float* xz    = (float*)alloc((size_t)ROWS * D_XZ    * sizeof(float));
    float* xc    = (float*)alloc((size_t)ROWS * D_INNER * sizeof(float));
    bf16*  xcb   = (bf16*) alloc((size_t)ROWS * D_INNER * sizeof(bf16));
    float* dbc   = (float*)alloc((size_t)ROWS * XPROJ_N * sizeof(float));
    bf16*  dlt   = (bf16*) alloc((size_t)ROWS * DT_PAD  * sizeof(bf16));
    float* Bm    = (float*)alloc((size_t)ROWS * D_STATE * sizeof(float));
    float* Cm    = (float*)alloc((size_t)ROWS * D_STATE * sizeof(float));
    float* delta = (float*)alloc((size_t)ROWS * D_INNER * sizeof(float));
    bf16*  yg    = (bf16*) alloc((size_t)ROWS * D_INNER * sizeof(bf16));
    float* uv[2];
    uv[0] = (float*)alloc((size_t)ROWS * D_MODEL * sizeof(float));
    uv[1] = (float*)alloc((size_t)ROWS * D_MODEL * sizeof(float));

    auto cdiv = [](size_t a, size_t b) { return (unsigned)((a + b - 1) / b); };

    for (int l = 0; l < 2; ++l) {
        int base = 1 + l * 10;
        const float* in_proj  = (const float*)d_in[base + 0];
        const float* conv_w   = (const float*)d_in[base + 1];
        const float* conv_b   = (const float*)d_in[base + 2];
        const float* x_proj   = (const float*)d_in[base + 3];
        const float* dt_w     = (const float*)d_in[base + 4];
        const float* dt_b     = (const float*)d_in[base + 5];
        const float* A_log    = (const float*)d_in[base + 6];
        const float* Dparam   = (const float*)d_in[base + 7];
        const float* out_proj = (const float*)d_in[base + 8];
        const float* norm_w   = (const float*)d_in[base + 9];

        // Weight conversion (f32 -> bf16, transposed to N x Kpad)
        convert_wt_kernel<<<cdiv((size_t)D_XZ * D_MODEL, 256), 256, 0, stream>>>(
            in_proj, wi_t, D_MODEL, D_XZ, D_MODEL);
        convert_wt_kernel<<<cdiv((size_t)XPROJ_N * D_INNER, 256), 256, 0, stream>>>(
            x_proj, wx_t, D_INNER, XPROJ_N, D_INNER);
        convert_wt_kernel<<<cdiv((size_t)D_INNER * DT_PAD, 256), 256, 0, stream>>>(
            dt_w, wd_t, DT_RANK, D_INNER, DT_PAD);
        convert_wt_kernel<<<cdiv((size_t)D_MODEL * D_INNER, 256), 256, 0, stream>>>(
            out_proj, wo_t, D_INNER, D_MODEL, D_INNER);

        // RMSNorm
        rmsnorm_kernel<<<ROWS, 256, 0, stream>>>(x, norm_w, xn);

        // GEMM1: xz = xn @ in_proj     (2048 x 3072, K=768) — LDS/async pipeline
        wmma_gemm_lds_kernel<<<(ROWS / 64) * (D_XZ / 64), 256, 0, stream>>>(
            xn, wi_t, xz, nullptr, ROWS, D_XZ, D_MODEL);

        // Causal depthwise conv + SiLU
        conv_silu_kernel<<<cdiv((size_t)ROWS * D_INNER, 256), 256, 0, stream>>>(
            xz, conv_w, conv_b, xc, xcb);

        // GEMM2: dbc = xc @ x_proj     (2048 x 80, K=1536) — direct fragments
        wmma_gemm_kernel<<<cdiv((size_t)(ROWS / 32) * (XPROJ_N / 16), 8), 256, 0, stream>>>(
            xcb, wx_t, dbc, nullptr, ROWS, XPROJ_N, D_INNER);

        // Split dbc -> dlt(bf16, padded) / Bm / Cm
        extract_dbc_kernel<<<cdiv((size_t)ROWS * 96, 256), 256, 0, stream>>>(
            dbc, dlt, Bm, Cm);

        // GEMM3: delta_pre = dlt @ dt_w   (2048 x 1536, K=64 padded)
        wmma_gemm_lds_kernel<<<(ROWS / 64) * (D_INNER / 64), 256, 0, stream>>>(
            dlt, wd_t, delta, nullptr, ROWS, D_INNER, DT_PAD);

        // delta = softplus(delta_pre + dt_b)
        delta_softplus_kernel<<<cdiv((size_t)ROWS * D_INNER, 256), 256, 0, stream>>>(
            delta, dt_b);

        // Selective scan + D*xc + SiLU(z) gating -> yg (bf16)
        selective_scan_kernel<<<BATCH * (D_INNER / 256), 256, 0, stream>>>(
            xz, xc, delta, Bm, Cm, A_log, Dparam, yg);

        // GEMM4: layer_out = yg @ out_proj + x   (2048 x 768, K=1536)
        wmma_gemm_lds_kernel<<<(ROWS / 64) * (D_MODEL / 64), 256, 0, stream>>>(
            yg, wo_t, uv[l], x, ROWS, D_MODEL, D_INNER);
    }

    // out = u * v
    final_mul_kernel<<<cdiv((size_t)ROWS * D_MODEL, 256), 256, 0, stream>>>(
        uv[0], uv[1], (float*)d_out);
}
